// DynFilter3D_6004364280342
// MI455X (gfx1250) — matrix operand order repaired and checked by reading, not compiled, not run on hardware
//
#include <hip/hip_runtime.h>

typedef __attribute__((ext_vector_type(16))) _Float16 v16h;
typedef __attribute__((ext_vector_type(8)))  float    v8f;

#define BB   8
#define TT   3
#define HH   192
#define WW   192
#define CO   16
#define KTOT 27                       // 3*3*3 taps, order (kh, kw, t)
#define NPIX (BB * HH * WW)

// One wave = one pixel per iteration:
//   D(16x16,f32) = A(16x32,f16: patch vector in every row)
//                * B(32x16,f16: this pixel's 27x16 filter, K padded to 32)
// Memory-bound (AI ~ 0.5 flop/B): f streams once with NT hints; x (3.5 MB)
// stays L2-resident. All inner code is branch-free so the 16 f-loads per
// lane issue back-to-back (full MLP) instead of load->wait->load.
__global__ __launch_bounds__(256) void DynFilter3D_wmma_kernel(
    const float* __restrict__ x,   // (B, T, H, W)
    const float* __restrict__ f,   // (B, H, W, 27, 16)
    float* __restrict__ out)       // (B, H, W, 16)
{
  const int lane   = threadIdx.x & 31;
  const int waveId = blockIdx.x * (blockDim.x >> 5) + (threadIdx.x >> 5);
  const int nWaves = gridDim.x * (blockDim.x >> 5);
  const int khalf  = lane >> 4;    // 0: lanes 0-15 (K 0-15), 1: lanes 16-31 (K 16-31)
  const int ncol   = lane & 15;    // output channel / B,D column

  for (int p = waveId; p < NPIX; p += nWaves) {
    // p = (b*H + h)*W + w  (wave-uniform)
    const int w  = p % WW;
    const int hw = p / WW;
    const int h  = hw % HH;
    const int b  = hw / HH;

    if (p + nWaves < NPIX)
      __builtin_prefetch(f + (size_t)(p + nWaves) * (KTOT * CO), 0, 0);

    // ================= B operand (32x16 f16) =================
    // lane (khalf,ncol): element j = f16(f[p, 16*khalf + j, ncol]), 0 for K>=27.
    // j=0..10 valid for both halves -> unconditional immediate-offset loads.
    // j=11..15 only invalid for khalf=1: clamp onto the k=26 cacheline
    // (already fetched -> no extra HBM traffic) and cndmask the value to 0.
    const float* fb = f + (size_t)p * (KTOT * CO) + khalf * (16 * CO) + ncol;
    float bf[16];
#pragma unroll
    for (int j = 0; j < 11; ++j)
      bf[j] = __builtin_nontemporal_load(fb + j * CO);
#pragma unroll
    for (int j = 11; j < 16; ++j) {
      const int jj = khalf ? 10 : j;                       // one cndmask
      float v = __builtin_nontemporal_load(fb + jj * CO);
      bf[j] = khalf ? 0.0f : v;                            // one cndmask
    }

    // ================= patches (27 values, k = (kh*3+kw)*3 + t) ==========
    // Wave-uniform interior/border split: interior (~98%) is 27 plain loads
    // at immediate offsets; border path clamps addresses and multiplies by
    // {0,1} validity (branch-free, zero padding semantics).
    float patch[KTOT];
    const bool interior = (h > 0) & (h < HH - 1) & (w > 0) & (w < WW - 1);
    if (interior) {
#pragma unroll
      for (int t = 0; t < 3; ++t)
#pragma unroll
        for (int kh = 0; kh < 3; ++kh) {
          const float* rp = x + (((size_t)b * TT + t) * HH + (h + kh - 1)) * WW + (w - 1);
#pragma unroll
          for (int kw2 = 0; kw2 < 3; ++kw2)
            patch[(kh * 3 + kw2) * 3 + t] = rp[kw2];
        }
    } else {
      const int   hr[3] = { h > 0 ? h - 1 : 0, h, h < HH - 1 ? h + 1 : HH - 1 };
      const int   wc[3] = { w > 0 ? w - 1 : 0, w, w < WW - 1 ? w + 1 : WW - 1 };
      const float vh[3] = { h > 0 ? 1.f : 0.f, 1.f, h < HH - 1 ? 1.f : 0.f };
      const float vw[3] = { w > 0 ? 1.f : 0.f, 1.f, w < WW - 1 ? 1.f : 0.f };
#pragma unroll
      for (int t = 0; t < 3; ++t)
#pragma unroll
        for (int kh = 0; kh < 3; ++kh) {
          const float* rp = x + (((size_t)b * TT + t) * HH + hr[kh]) * WW;
#pragma unroll
          for (int kw2 = 0; kw2 < 3; ++kw2)
            patch[(kh * 3 + kw2) * 3 + t] = vh[kh] * vw[kw2] * rp[wc[kw2]];
        }
    }

    // ================= A operand (16x32 f16) =================
    // Contiguous K order (same as B), element e of lane:
//   khalf=0: K = e                    (0..15)
//   khalf=1: K = 16 + e               (16..31; >=27 -> 0)
    // Every row identical (patch vector) -> D rows identical, row 0 = result.
    v16h am;
#pragma unroll
    for (int e = 0; e < 16; ++e) {
      const int klo = e;                         // <= 15, compile-time
      const int khi = klo + 16;                  // <= 31, compile-time
      const float plo = patch[klo];
      const float phi = (khi < KTOT) ? patch[khi] : 0.0f;   // folds at compile time
      am[e] = (_Float16)(khalf ? phi : plo);     // one cndmask + cvt
    }

    v16h bm;
#pragma unroll
    for (int e = 0; e < 16; ++e)
      bm[e] = (_Float16)bf[e];

    // D = A*B + 0, f32 accumulate
    v8f c = {};
    c = __builtin_amdgcn_wmma_f32_16x16x32_f16(
        /*neg_a=*/false, am, /*neg_b=*/false, bm,
        /*c_mod=*/(short)0, c, /*reuse_a=*/false, /*reuse_b=*/false);

    // D row 0 lives in VGPR0 of lanes 0..15 (M=0, N=lane)
    if (lane < 16)
      __builtin_nontemporal_store(c[0], out + (size_t)p * CO + lane);
  }
}

extern "C" void kernel_launch(void* const* d_in, const int* in_sizes, int n_in,
                              void* d_out, int out_size, void* d_ws, size_t ws_size,
                              hipStream_t stream) {
  (void)in_sizes; (void)n_in; (void)out_size; (void)d_ws; (void)ws_size;
  const float* x = (const float*)d_in[0];
  const float* f = (const float*)d_in[1];
  float* out     = (float*)d_out;

  // 2048 blocks x 256 threads = 16384 wave32 waves -> ~18 pixels/wave;
  // deep oversubscription to hide the f streaming latency.
  dim3 grid(2048), block(256);
  hipLaunchKernelGGL(DynFilter3D_wmma_kernel, grid, block, 0, stream, x, f, out);
}
